// MambaBlock_8280696947370
// MI455X (gfx1250) — matrix operand order repaired and checked by reading, not compiled, not run on hardware
//
#include <hip/hip_runtime.h>
#include <hip/hip_bf16.h>
#include <stdint.h>

// ---------------------------------------------------------------------------
// MambaBlock for MI455X (gfx1250, wave32, WMMA).
//  - 5 dense GEMMs (~97 GFLOP) via v_wmma_f32_16x16x32_f16 (f32 accumulate).
//  - fp32 chunked scan for h_t = h_{t-1}@A + v_t (dense A) to preserve the
//    recurrence numerics while exposing 128-way parallelism.
//  - Wave tile 32x64 (2x4 WMMA accum) => ~21 FLOP/byte from cache in the
//    main loop; block tile 64x256 with 8 waves per WG.
// ---------------------------------------------------------------------------

typedef __attribute__((ext_vector_type(16))) _Float16 v16h;
typedef __attribute__((ext_vector_type(8)))  _Float16 v8h;
typedef __attribute__((ext_vector_type(8)))  float    v8f;

#define WMMA_F16(a, b, c) \
  __builtin_amdgcn_wmma_f32_16x16x32_f16(false, (a), false, (b), (short)0, (c), false, false)

#define BATCH   8
#define TLEN    2048
#define DMODEL  1024
#define DSTATE  256
#define MROWS   (BATCH * TLEN)   // 16384
#define CHUNK   128
#define NCHUNK  (TLEN / CHUNK)   // 16

// ===========================================================================
// Conversion / layout kernels
// ===========================================================================

__global__ void __launch_bounds__(256) f32_to_f16_kernel(const float* __restrict__ src,
                                                         _Float16* __restrict__ dst,
                                                         size_t n) {
  size_t i = (size_t)blockIdx.x * 256 + threadIdx.x;
  size_t stride = (size_t)gridDim.x * 256;
  for (; i < n; i += stride) dst[i] = (_Float16)src[i];
}

// src is [K][N] row-major fp32 (Flax kernel); dst is [N][K] row-major f16.
__global__ void __launch_bounds__(256) transpose_to_f16_kernel(const float* __restrict__ src,
                                                               _Float16* __restrict__ dst,
                                                               int K, int N) {
  size_t i = (size_t)blockIdx.x * 256 + threadIdx.x;
  size_t n = (size_t)K * N;
  size_t stride = (size_t)gridDim.x * 256;
  for (; i < n; i += stride) {
    int k = (int)(i / N);
    int c = (int)(i % N);
    dst[(size_t)c * K + k] = (_Float16)src[i];
  }
}

// C,D are [DMODEL][DSTATE]; pack rows as [C_row | D_row] -> [DMODEL][2*DSTATE] f16.
__global__ void __launch_bounds__(256) pack_cd_kernel(const float* __restrict__ C,
                                                      const float* __restrict__ D,
                                                      _Float16* __restrict__ CDt) {
  size_t i = (size_t)blockIdx.x * 256 + threadIdx.x;
  size_t n = (size_t)DMODEL * DSTATE;
  if (i >= n) return;
  int d = (int)(i >> 8);       // DSTATE == 256
  int s = (int)(i & 255);
  CDt[(size_t)d * (2 * DSTATE) + s]          = (_Float16)C[i];
  CDt[(size_t)d * (2 * DSTATE) + DSTATE + s] = (_Float16)D[i];
}

// ===========================================================================
// WMMA f16 GEMM:  out[m][n] = act( sum_k A[m][k] * Bt[n][k] + bias[n] )
//   A  : [M][lda] f16   Bt : [N][ldb] f16 (pre-transposed weights)
//   MODE 0: fp32 out (+bias if non-null)
//   MODE 1: sigmoid -> f16 out
//   MODE 2: f16 out (+bias)
//   MODE 3: multiply by gate (f16) -> f16 out
// Block tile 64x256, 8 waves (2x4), each wave 32x64 = 2x4 WMMA tiles.
// ===========================================================================
template <int MODE>
__global__ void __launch_bounds__(256)
gemm_wmma_f16(const _Float16* __restrict__ A, int lda,
              const _Float16* __restrict__ Bt, int ldb,
              const float* __restrict__ bias,
              const _Float16* __restrict__ gate, int ldg,
              void* __restrict__ outp, int ldo,
              int M, int N, int K) {
  (void)M; (void)N;
  const int lane = threadIdx.x & 31;
  const int wave = threadIdx.x >> 5;
  const int wm   = wave >> 2;            // 0..1
  const int wn   = wave & 3;             // 0..3
  const int m0   = blockIdx.y * 64 + wm * 32;
  const int n0   = blockIdx.x * 256 + wn * 64;
  const int lr   = lane & 15;
  const int hi   = lane >> 4;            // 0 or 1 (half-wave)

  // A fragment (code's own k order, not the ISA table's): lane half picks contiguous K 0..15 / 16..31
  const _Float16* a0 = A + (size_t)(m0 + lr) * lda + hi * 16;
  const _Float16* a1 = a0 + (size_t)16 * lda;
  // B fragment (code's own k order, not the ISA table's): lane half picks contiguous K 0..15 / 16..31
  const _Float16* b0 = Bt + (size_t)(n0 + lr) * ldb + hi * 16;
  const _Float16* b1 = b0 + (size_t)16 * ldb;
  const _Float16* b2 = b0 + (size_t)32 * ldb;
  const _Float16* b3 = b0 + (size_t)48 * ldb;

  v8f acc[2][4];
#pragma unroll
  for (int i = 0; i < 2; ++i)
#pragma unroll
    for (int j = 0; j < 4; ++j) acc[i][j] = (v8f){};

  for (int k0 = 0; k0 < K; k0 += 32) {
    v8h alo0 = *(const v8h*)(a0 + k0);
    v8h ahi0 = *(const v8h*)(a0 + k0 + 8);
    v8h alo1 = *(const v8h*)(a1 + k0);
    v8h ahi1 = *(const v8h*)(a1 + k0 + 8);
    v16h fa0 = __builtin_shufflevector(alo0, ahi0, 0,1,2,3,4,5,6,7,8,9,10,11,12,13,14,15);
    v16h fa1 = __builtin_shufflevector(alo1, ahi1, 0,1,2,3,4,5,6,7,8,9,10,11,12,13,14,15);
    v16h fb0 = *(const v16h*)(b0 + k0);
    v16h fb1 = *(const v16h*)(b1 + k0);
    v16h fb2 = *(const v16h*)(b2 + k0);
    v16h fb3 = *(const v16h*)(b3 + k0);
    acc[0][0] = WMMA_F16(fa0, fb0, acc[0][0]);
    acc[0][1] = WMMA_F16(fa0, fb1, acc[0][1]);
    acc[0][2] = WMMA_F16(fa0, fb2, acc[0][2]);
    acc[0][3] = WMMA_F16(fa0, fb3, acc[0][3]);
    acc[1][0] = WMMA_F16(fa1, fb0, acc[1][0]);
    acc[1][1] = WMMA_F16(fa1, fb1, acc[1][1]);
    acc[1][2] = WMMA_F16(fa1, fb2, acc[1][2]);
    acc[1][3] = WMMA_F16(fa1, fb3, acc[1][3]);
  }

  auto emit = [&](v8f a, int mt, int nt) {
    const int col = n0 + nt + lr;
    float bv = 0.f;
    if constexpr (MODE == 0 || MODE == 2) {
      if (bias) bv = bias[col];
    }
#pragma unroll
    for (int r = 0; r < 8; ++r) {
      const int row = m0 + mt + hi * 8 + r;
      float t = a[r] + bv;
      if constexpr (MODE == 0) {
        ((float*)outp)[(size_t)row * ldo + col] = t;
      } else if constexpr (MODE == 1) {
        t = 1.0f / (1.0f + __expf(-t));
        ((_Float16*)outp)[(size_t)row * ldo + col] = (_Float16)t;
      } else if constexpr (MODE == 2) {
        ((_Float16*)outp)[(size_t)row * ldo + col] = (_Float16)t;
      } else {  // MODE == 3
        t *= (float)gate[(size_t)row * ldg + col];
        ((_Float16*)outp)[(size_t)row * ldo + col] = (_Float16)t;
      }
    }
  };
#pragma unroll
  for (int i = 0; i < 2; ++i)
#pragma unroll
    for (int j = 0; j < 4; ++j) emit(acc[i][j], i * 16, j * 16);
}

// ===========================================================================
// fp32 scan machinery
// ===========================================================================

// Z = X @ Y for 256x256 fp32 (used for repeated squaring of A -> A^128).
__global__ void __launch_bounds__(256) matmul256_kernel(const float* __restrict__ X,
                                                        const float* __restrict__ Y,
                                                        float* __restrict__ Z) {
  const int col = threadIdx.x;
  const int row = blockIdx.x;
  float acc = 0.f;
#pragma unroll 8
  for (int k = 0; k < DSTATE; ++k)
    acc = fmaf(X[(size_t)row * DSTATE + k], Y[(size_t)k * DSTATE + col], acc);
  Z[(size_t)row * DSTATE + col] = acc;
}

// Phase 1: chunk-local scan p_j = p_{j-1}@A + v_j starting from 0, in-place on v.
// One WG per (chunk, batch); A staged in 256KB LDS (<= 320KB/WGP on CDNA5).
__global__ void __launch_bounds__(256) local_scan_kernel(const float* __restrict__ Amat,
                                                         float* __restrict__ v,
                                                         float* __restrict__ p_last) {
  extern __shared__ float smem[];
  float* As = smem;              // 256*256 floats
  float* h  = smem + DSTATE * DSTATE;
  const int n = threadIdx.x;
  const int c = blockIdx.x;
  const int b = blockIdx.y;
  for (int i = n; i < DSTATE * DSTATE; i += 256) As[i] = Amat[i];
  h[n] = 0.f;
  __syncthreads();
  float* vbase = v + ((size_t)b * TLEN + (size_t)c * CHUNK) * DSTATE;
  for (int j = 0; j < CHUNK; ++j) {
    float acc = vbase[(size_t)j * DSTATE + n];
#pragma unroll 8
    for (int k = 0; k < DSTATE; ++k)
      acc = fmaf(h[k], As[k * DSTATE + n], acc);
    __syncthreads();
    h[n] = acc;
    vbase[(size_t)j * DSTATE + n] = acc;
    __syncthreads();
  }
  p_last[((size_t)c * BATCH + b) * DSTATE + n] = h[n];
}

// Phase 2: sequential carry across chunks: S_c = S_{c-1}@A^CHUNK + p_last[c].
// One WG; H_state[c] = true state at end of chunk c.
__global__ void __launch_bounds__(256) carry_scan_kernel(const float* __restrict__ A_L,
                                                         const float* __restrict__ p_last,
                                                         float* __restrict__ H_state) {
  extern __shared__ float smem[];
  float* As = smem;                    // 256*256
  float* S  = smem + DSTATE * DSTATE;  // 8*256
  const int n = threadIdx.x;
  for (int i = n; i < DSTATE * DSTATE; i += 256) As[i] = A_L[i];
  for (int b = 0; b < BATCH; ++b) S[b * DSTATE + n] = 0.f;
  __syncthreads();
  for (int c = 0; c < NCHUNK; ++c) {
    float acc[BATCH];
    for (int b = 0; b < BATCH; ++b) {
      float a = p_last[((size_t)c * BATCH + b) * DSTATE + n];
#pragma unroll 8
      for (int k = 0; k < DSTATE; ++k)
        a = fmaf(S[b * DSTATE + k], As[k * DSTATE + n], a);
      acc[b] = a;
    }
    __syncthreads();
    for (int b = 0; b < BATCH; ++b) {
      S[b * DSTATE + n] = acc[b];
      H_state[((size_t)c * BATCH + b) * DSTATE + n] = acc[b];
    }
    __syncthreads();
  }
}

// Phase 3: fix-up hs_j = H_{c-1}@A^{j+1} + p_j; write f16 into left half of [hs|u].
__global__ void __launch_bounds__(256) fixup_scan_kernel(const float* __restrict__ Amat,
                                                         const float* __restrict__ H_state,
                                                         const float* __restrict__ p,
                                                         _Float16* __restrict__ hu) {
  extern __shared__ float smem[];
  float* As = smem;
  float* q  = smem + DSTATE * DSTATE;
  const int n = threadIdx.x;
  const int c = blockIdx.x;
  const int b = blockIdx.y;
  const float* pbase = p + ((size_t)b * TLEN + (size_t)c * CHUNK) * DSTATE;
  _Float16* hubase = hu + ((size_t)b * TLEN + (size_t)c * CHUNK) * (2 * DSTATE);
  if (c == 0) {  // no incoming state: hs == p
    for (int j = 0; j < CHUNK; ++j)
      hubase[(size_t)j * (2 * DSTATE) + n] = (_Float16)pbase[(size_t)j * DSTATE + n];
    return;
  }
  for (int i = n; i < DSTATE * DSTATE; i += 256) As[i] = Amat[i];
  q[n] = H_state[((size_t)(c - 1) * BATCH + b) * DSTATE + n];
  __syncthreads();
  for (int j = 0; j < CHUNK; ++j) {
    float acc = 0.f;
#pragma unroll 8
    for (int k = 0; k < DSTATE; ++k)
      acc = fmaf(q[k], As[k * DSTATE + n], acc);
    __syncthreads();
    q[n] = acc;
    hubase[(size_t)j * (2 * DSTATE) + n] =
        (_Float16)(acc + pbase[(size_t)j * DSTATE + n]);
    __syncthreads();
  }
}

// ===========================================================================
// Host launcher
// ===========================================================================
extern "C" void kernel_launch(void* const* d_in, const int* in_sizes, int n_in,
                              void* d_out, int out_size, void* d_ws, size_t ws_size,
                              hipStream_t stream) {
  (void)in_sizes; (void)n_in; (void)out_size; (void)ws_size;

  const float* x      = (const float*)d_in[0];
  const float* W_in   = (const float*)d_in[1];
  const float* b_in   = (const float*)d_in[2];
  const float* W_gate = (const float*)d_in[3];
  const float* b_gate = (const float*)d_in[4];
  const float* W_out  = (const float*)d_in[5];
  const float* b_out  = (const float*)d_in[6];
  const float* Amat   = (const float*)d_in[7];
  const float* Bm     = (const float*)d_in[8];
  const float* Cm     = (const float*)d_in[9];
  const float* Dm     = (const float*)d_in[10];
  float* out = (float*)d_out;

  // ---- workspace carve-up (~141 MB) ----
  uint8_t* wsb = (uint8_t*)d_ws;
  size_t off = 0;
  auto take = [&](size_t bytes) -> void* {
    void* p = wsb + off;
    off += (bytes + 255) & ~(size_t)255;
    return p;
  };
  _Float16* xh     = (_Float16*)take((size_t)MROWS * DMODEL * 2);        // x in f16
  _Float16* Wint   = (_Float16*)take((size_t)DSTATE * DMODEL * 2);       // W_in^T
  _Float16* Wgatet = (_Float16*)take((size_t)DMODEL * DMODEL * 2);       // W_gate^T
  _Float16* Woutt  = (_Float16*)take((size_t)DMODEL * DMODEL * 2);       // W_out^T
  _Float16* Bmt    = (_Float16*)take((size_t)DSTATE * DSTATE * 2);       // Bm^T
  _Float16* CDt    = (_Float16*)take((size_t)DMODEL * 2 * DSTATE * 2);   // [C|D] rows
  _Float16* hu     = (_Float16*)take((size_t)MROWS * 2 * DSTATE * 2);    // [hs|u] f16
  _Float16* gateh  = (_Float16*)take((size_t)MROWS * DMODEL * 2);        // sigmoid gate f16
  float*    vbuf   = (float*)take((size_t)MROWS * DSTATE * 4);           // v, then p (in place)
  _Float16* zg     = (_Float16*)take((size_t)MROWS * DMODEL * 2);        // gated readout f16
  float*    p_last = (float*)take((size_t)NCHUNK * BATCH * DSTATE * 4);
  float*    Hstate = (float*)take((size_t)NCHUNK * BATCH * DSTATE * 4);
  float*    apow0  = (float*)take((size_t)DSTATE * DSTATE * 4);
  float*    apow1  = (float*)take((size_t)DSTATE * DSTATE * 4);

  const dim3 blk(256);
  const size_t LDS_SCAN  = (size_t)(DSTATE * DSTATE + DSTATE) * 4;          // 257 KB
  const size_t LDS_CARRY = (size_t)(DSTATE * DSTATE + BATCH * DSTATE) * 4;  // 264 KB

  // ---- layout conversions ----
  f32_to_f16_kernel<<<4096, blk, 0, stream>>>(x, xh, (size_t)MROWS * DMODEL);
  transpose_to_f16_kernel<<<1024, blk, 0, stream>>>(W_in,   Wint,   DMODEL, DSTATE);
  transpose_to_f16_kernel<<<4096, blk, 0, stream>>>(W_gate, Wgatet, DMODEL, DMODEL);
  transpose_to_f16_kernel<<<4096, blk, 0, stream>>>(W_out,  Woutt,  DMODEL, DMODEL);
  transpose_to_f16_kernel<<<256,  blk, 0, stream>>>(Bm,     Bmt,    DSTATE, DSTATE);
  pack_cd_kernel<<<(DMODEL * DSTATE) / 256, blk, 0, stream>>>(Cm, Dm, CDt);

  // ---- u = x@W_in + b_in  (f16, into right half of [hs|u]) ----
  gemm_wmma_f16<2><<<dim3(DSTATE / 256, MROWS / 64), blk, 0, stream>>>(
      xh, DMODEL, Wint, DMODEL, b_in, nullptr, 0, hu + DSTATE, 2 * DSTATE,
      MROWS, DSTATE, DMODEL);

  // ---- gate = sigmoid(x@W_gate) (f16; b_gate passed but not added) ----
  gemm_wmma_f16<1><<<dim3(DMODEL / 256, MROWS / 64), blk, 0, stream>>>(
      xh, DMODEL, Wgatet, DMODEL, b_gate, nullptr, 0, gateh, DMODEL,
      MROWS, DMODEL, DMODEL);

  // ---- v = u@Bm (fp32) ----
  gemm_wmma_f16<0><<<dim3(DSTATE / 256, MROWS / 64), blk, 0, stream>>>(
      hu + DSTATE, 2 * DSTATE, Bmt, DSTATE, nullptr, nullptr, 0, vbuf, DSTATE,
      MROWS, DSTATE, DSTATE);

  // ---- A^CHUNK by repeated squaring (7 launches -> apow0 holds A^128) ----
  matmul256_kernel<<<DSTATE, blk, 0, stream>>>(Amat,  Amat,  apow0);  // A^2
  matmul256_kernel<<<DSTATE, blk, 0, stream>>>(apow0, apow0, apow1);  // A^4
  matmul256_kernel<<<DSTATE, blk, 0, stream>>>(apow1, apow1, apow0);  // A^8
  matmul256_kernel<<<DSTATE, blk, 0, stream>>>(apow0, apow0, apow1);  // A^16
  matmul256_kernel<<<DSTATE, blk, 0, stream>>>(apow1, apow1, apow0);  // A^32
  matmul256_kernel<<<DSTATE, blk, 0, stream>>>(apow0, apow0, apow1);  // A^64
  matmul256_kernel<<<DSTATE, blk, 0, stream>>>(apow1, apow1, apow0);  // A^128

  // ---- 3-phase chunked scan (fp32) ----
  local_scan_kernel<<<dim3(NCHUNK, BATCH), blk, LDS_SCAN, stream>>>(Amat, vbuf, p_last);
  carry_scan_kernel<<<1, blk, LDS_CARRY, stream>>>(apow0, p_last, Hstate);
  fixup_scan_kernel<<<dim3(NCHUNK, BATCH), blk, LDS_SCAN, stream>>>(Amat, Hstate, vbuf, hu);

  // ---- z = (hs@C^T + u@D^T) * gate (f16) ----
  gemm_wmma_f16<3><<<dim3(DMODEL / 256, MROWS / 64), blk, 0, stream>>>(
      hu, 2 * DSTATE, CDt, 2 * DSTATE, nullptr, gateh, DMODEL, zg, DMODEL,
      MROWS, DMODEL, 2 * DSTATE);

  // ---- out = z@W_out + b_out (fp32) ----
  gemm_wmma_f16<0><<<dim3(DMODEL / 256, MROWS / 64), blk, 0, stream>>>(
      zg, DMODEL, Woutt, DMODEL, b_out, nullptr, 0, out, DMODEL,
      MROWS, DMODEL, DMODEL);
}
